// _GATLayer_36859409335029
// MI455X (gfx1250) — compile-verified, matrix operand order checked by reading, not run on hardware
//
#include <hip/hip_runtime.h>
#include <hip/hip_bf16.h>

#define V_NODES 8192
#define E_EDGES 262144
#define DIM     128
#define NEG_SLOPE 0.2f

typedef __attribute__((ext_vector_type(16))) _Float16 v16h;
typedef __attribute__((ext_vector_type(8)))  _Float16 v8h;
typedef __attribute__((ext_vector_type(8)))  float    v8f;

// ---------- f32 -> f16 cast ----------
__global__ void k_cast_f32_to_f16(const float* __restrict__ src,
                                  _Float16* __restrict__ dst, int n) {
  int i = blockIdx.x * blockDim.x + threadIdx.x;
  if (i < n) dst[i] = (_Float16)src[i];
}

// ---------- hw = h @ W^T via WMMA f16->f32 ----------
// Block = 256 threads = 8 waves. Block owns 16 rows of h; wave w owns the
// 16-column tile n0 = 16*w. A tile (16x128 halves) staged in LDS, shared by
// all 8 waves. B fragment reads row-major W16 directly (B[k][n] = W[n][k]).
__global__ void k_gemm_hw(const _Float16* __restrict__ h16,
                          const _Float16* __restrict__ w16,
                          float* __restrict__ hw) {
  __shared__ _Float16 Alds[16][136];  // pad 128->136 halves: dodge bank conflicts
  const int row0 = blockIdx.x * 16;
  const int tid  = threadIdx.x;

  // cooperative stage of A: 16 rows x 128 halves = 256 threads x 16B
  {
    const int r = tid >> 4;
    const int c = (tid & 15) * 8;
    *(v8h*)&Alds[r][c] = *(const v8h*)&h16[(size_t)(row0 + r) * DIM + c];
  }
  __syncthreads();

  const int wave = tid >> 5;        // wave32
  const int lane = tid & 31;
  const int n0   = wave * 16;
  const int ml   = lane & 15;       // row (A) / col (B) within tile
  const int kh   = (lane >> 4) * 8; // half-wave K offset per ISA layout

  const _Float16* wrow = &w16[(size_t)(n0 + ml) * DIM];
  v8f c = {};
#pragma unroll
  for (int k0 = 0; k0 < DIM; k0 += 32) {
    v8h a0 = *(const v8h*)&Alds[ml][k0 + kh];
    v8h a1 = *(const v8h*)&Alds[ml][k0 + 16 + kh];
    v8h b0 = *(const v8h*)&wrow[k0 + kh];
    v8h b1 = *(const v8h*)&wrow[k0 + 16 + kh];
    v16h a = __builtin_shufflevector(a0, a1, 0,1,2,3,4,5,6,7,8,9,10,11,12,13,14,15);
    v16h b = __builtin_shufflevector(b0, b1, 0,1,2,3,4,5,6,7,8,9,10,11,12,13,14,15);
    c = __builtin_amdgcn_wmma_f32_16x16x32_f16(false, a, false, b,
                                               (short)0, c, false, false);
  }

  // C layout: VGPR r -> M = r + 8*(lane>=16), N = lane&15
  const int mbase = row0 + 8 * (lane >> 4);
#pragma unroll
  for (int r = 0; r < 8; ++r) {
    hw[(size_t)(mbase + r) * DIM + n0 + ml] = c[r];
  }
}

// ---------- per-vertex scores s1,s2 and column sum of hw ----------
__global__ void k_scores(const float* __restrict__ hw,
                         const float* __restrict__ att_w,
                         float* __restrict__ s1, float* __restrict__ s2,
                         float* __restrict__ stot) {
  __shared__ float r1[DIM];
  __shared__ float r2[DIM];
  const int v = blockIdx.x;
  const int t = threadIdx.x;  // 128 threads
  float x = hw[(size_t)v * DIM + t];
  atomicAdd(&stot[t], x);                 // Sum_v hw[v][t]
  r1[t] = x * att_w[t];
  r2[t] = x * att_w[DIM + t];
  __syncthreads();
  for (int s = 64; s > 0; s >>= 1) {
    if (t < s) { r1[t] += r1[t + s]; r2[t] += r2[t + s]; }
    __syncthreads();
  }
  if (t == 0) { s1[v] = r1[0]; s2[v] = r2[0]; }
}

// ---------- pass 1 over edges: row max of max(a, 0) ----------
// scores clamped at >=0 (softmax row always contains zeros), so int-compare
// atomicMax on the float bit pattern is order-preserving.
__global__ void k_edge_max(const long long* __restrict__ ei,
                           const float* __restrict__ s1,
                           const float* __restrict__ s2,
                           int* __restrict__ rowmax) {
  int e = blockIdx.x * blockDim.x + threadIdx.x;
  if (e >= E_EDGES) return;
  int src = (int)ei[e];
  int dst = (int)ei[E_EDGES + e];
  float a = s1[src] + s2[dst];
  a = a > 0.f ? a : NEG_SLOPE * a;
  a = fmaxf(a, 0.f);
  atomicMax(&rowmax[src], __float_as_int(a));
}

// ---------- em[r] = exp(-rowmax[r]) ----------
__global__ void k_rowprep(const int* __restrict__ rowmax, float* __restrict__ em) {
  int r = blockIdx.x * blockDim.x + threadIdx.x;
  if (r < V_NODES) em[r] = __expf(-__int_as_float(rowmax[r]));
}

// ---------- pass 2 over edges: weighted scatter into out ----------
// 128 threads per edge (2 edges per 256-thread block), one feature per thread.
__global__ void k_edge_accum(const long long* __restrict__ ei,
                             const float* __restrict__ s1,
                             const float* __restrict__ s2,
                             const int* __restrict__ rowmax,
                             const float* __restrict__ em,
                             const float* __restrict__ hw,
                             float* __restrict__ sumexp,
                             int* __restrict__ cnt,
                             float* __restrict__ out) {
  int e = blockIdx.x * 2 + (threadIdx.x >> 7);
  int t = threadIdx.x & 127;
  int src = (int)ei[e];
  int dst = (int)ei[E_EDGES + e];
  float a = s1[src] + s2[dst];
  a = a > 0.f ? a : NEG_SLOPE * a;
  float w = __expf(a - __int_as_float(rowmax[src]));
  if (t == 0) {
    atomicAdd(&sumexp[src], w);
    atomicAdd(&cnt[src], 1);
  }
  float coef = w - em[src];  // replace the implicit exp(-m) of a non-edge col
  atomicAdd(&out[(size_t)src * DIM + t], coef * hw[(size_t)dst * DIM + t]);
}

// ---------- finalize: add non-edge mass, divide by denominator ----------
__global__ void k_finalize(const float* __restrict__ em,
                           const float* __restrict__ sumexp,
                           const int* __restrict__ cnt,
                           const float* __restrict__ stot,
                           float* __restrict__ out) {
  int i = blockIdx.x * blockDim.x + threadIdx.x;
  if (i >= V_NODES * DIM) return;
  int r = i >> 7;
  int t = i & 127;
  float D = (float)(V_NODES - cnt[r]) * em[r] + sumexp[r];
  out[i] = (out[i] + em[r] * stot[t]) / D;
}

extern "C" void kernel_launch(void* const* d_in, const int* in_sizes, int n_in,
                              void* d_out, int out_size, void* d_ws, size_t ws_size,
                              hipStream_t stream) {
  (void)in_sizes; (void)n_in; (void)out_size; (void)ws_size;
  const float*     h   = (const float*)d_in[0];
  const long long* ei  = (const long long*)d_in[1];  // int64 (2, E)
  const float*     W   = (const float*)d_in[2];
  const float*     att = (const float*)d_in[3];
  float*           out = (float*)d_out;

  char* ws = (char*)d_ws;
  _Float16* h16 = (_Float16*)ws; ws += (size_t)V_NODES * DIM * 2;
  _Float16* w16 = (_Float16*)ws; ws += (size_t)DIM * DIM * 2;
  float* hw     = (float*)ws;    ws += (size_t)V_NODES * DIM * 4;
  float* s1     = (float*)ws;    ws += (size_t)V_NODES * 4;
  float* s2     = (float*)ws;    ws += (size_t)V_NODES * 4;
  int*   rowmax = (int*)ws;      ws += (size_t)V_NODES * 4;
  float* em     = (float*)ws;    ws += (size_t)V_NODES * 4;
  float* sumexp = (float*)ws;    ws += (size_t)V_NODES * 4;
  int*   cnt    = (int*)ws;      ws += (size_t)V_NODES * 4;
  float* stot   = (float*)ws;    ws += (size_t)DIM * 4;

  // accumulators must start at zero every call (graph-replay safe)
  hipMemsetAsync(out,    0, (size_t)V_NODES * DIM * 4, stream);
  hipMemsetAsync(rowmax, 0, (size_t)V_NODES * 4, stream);
  hipMemsetAsync(sumexp, 0, (size_t)V_NODES * 4, stream);
  hipMemsetAsync(cnt,    0, (size_t)V_NODES * 4, stream);
  hipMemsetAsync(stot,   0, (size_t)DIM * 4, stream);

  k_cast_f32_to_f16<<<(V_NODES * DIM + 255) / 256, 256, 0, stream>>>(h, h16, V_NODES * DIM);
  k_cast_f32_to_f16<<<(DIM * DIM + 255) / 256, 256, 0, stream>>>(W, w16, DIM * DIM);

  k_gemm_hw<<<V_NODES / 16, 256, 0, stream>>>(h16, w16, hw);

  k_scores<<<V_NODES, DIM, 0, stream>>>(hw, att, s1, s2, stot);

  k_edge_max<<<(E_EDGES + 255) / 256, 256, 0, stream>>>(ei, s1, s2, rowmax);
  k_rowprep<<<(V_NODES + 255) / 256, 256, 0, stream>>>(rowmax, em);
  k_edge_accum<<<E_EDGES / 2, 256, 0, stream>>>(ei, s1, s2, rowmax, em, hw, sumexp, cnt, out);

  k_finalize<<<(V_NODES * DIM + 255) / 256, 256, 0, stream>>>(em, sumexp, cnt, stot, out);
}
